// SurfNNConvAutoencoder_40999757808034
// MI455X (gfx1250) — matrix operand order repaired and checked by reading, not compiled, not run on hardware
//
#include <hip/hip_runtime.h>

#define DI __device__ __forceinline__
typedef unsigned int u32;
typedef unsigned short u16;

typedef __attribute__((ext_vector_type(16))) __bf16 v16bf;
typedef __attribute__((ext_vector_type(8)))  float  v8f;
typedef __attribute__((ext_vector_type(8)))  u32    v8u;

#define NN 50000
#define NE 200000
#define WPB_E 4   // waves per block, edge kernel:  12500 tiles = 3125 * 4
#define WPB_N 5   // waves per block, node kernels:  3125 tiles =  625 * 5
static_assert(NN % 16 == 0 && (NN / 16) % WPB_N == 0, "exact node tiling");
static_assert(NE % 16 == 0 && (NE / 16) % WPB_E == 0, "exact edge tiling");

// ---------- helpers ----------
DI v8f wmma_bf16(v8u a, v8u b, v8f c) {
  return __builtin_amdgcn_wmma_f32_16x16x32_bf16(
      false, __builtin_bit_cast(v16bf, a),
      false, __builtin_bit_cast(v16bf, b),
      (short)0, c, false, false);
}

// B fragments prepacked as [frag][lane][16 halfs]; one 32B vector per lane.
DI v8u load_frag(const u16* __restrict__ f, int idx) {
  return *((const v8u*)(f + idx * 512) + (threadIdx.x & 31));
}

DI u32 pack2_bf16(float lo, float hi) {  // truncating f32->bf16 pair
  return (__float_as_uint(lo) >> 16) | (__float_as_uint(hi) & 0xFFFF0000u);
}

DI u16 bf16_rne(float f) {
  u32 x = __float_as_uint(f);
  return (u16)((x + 0x7FFFu + ((x >> 16) & 1u)) >> 16);
}

DI u16 bf16_trunc(float f) { return (u16)(__float_as_uint(f) >> 16); }

DI u32 pk_mul_bf16(u32 a, u32 b) {
  u32 d;
  asm volatile("v_pk_mul_bf16 %0, %1, %2" : "=v"(d) : "v"(a), "v"(b));
  return d;
}

// Build 16x32 bf16 A-fragment from a contiguous 32-half row (bf16 in memory).
// lane<16: dwords {0..3, 8..11}; lane>=16: {4..7, 12..15}.
DI v8u afrag_row32(const u32* row, int hi) {
  v8u a;
#pragma unroll
  for (int p = 0; p < 4; ++p) {
    a[p]     = row[hi * 4 + p];
    a[p + 4] = row[hi * 4 + 8 + p];
  }
  return a;
}

// ---------- small utility kernels ----------
__global__ void k_cvt_bf16(const float* __restrict__ x, u16* __restrict__ o, int n) {
  int t = blockIdx.x * 256 + threadIdx.x;
  if (t < n) o[t] = bf16_rne(x[t]);
}

// Pre-pack B fragments. mode 0: plain row-major W[Ksrc x N] (zero pad K).
// mode 1: W2R mapping from edge w2[32,1024]: B[k,n] = w2[k>>5, ((k&31)<<5)+n].
__global__ void k_prepack(const float* __restrict__ W, u16* __restrict__ out,
                          int ksteps, int ntiles, int Ksrc, int N, int mode) {
  int t = blockIdx.x * 256 + threadIdx.x;
  int total = ksteps * ntiles * 512;
  if (t >= total) return;
  int frag = t >> 9, r = t & 511;
  int lane = r >> 4, i = r & 15;
  int kstep = frag / ntiles, nt = frag - kstep * ntiles;
  int n = nt * 16 + (lane & 15);
  int k = kstep * 32 + ((i < 8) ? ((lane >> 4) << 3) + i : 16 + ((lane >> 4) << 3) + (i - 8));
  float v = 0.f;
  if (mode == 0) { if (k < Ksrc) v = W[k * N + n]; }
  else           { v = W[(k >> 5) * 1024 + ((k & 31) << 5) + n]; }
  out[t] = bf16_rne(v);
}

__global__ void k_zero(float* __restrict__ agg, float* __restrict__ cnt, int na, int nc) {
  int t = blockIdx.x * 256 + threadIdx.x;
  if (t < na) agg[t] = 0.f;
  if (t < nc) cnt[t] = 0.f;
}

__global__ void k_count(const int* __restrict__ eidx, float* __restrict__ cnt, int nE) {
  int e = blockIdx.x * 256 + threadIdx.x;
  if (e < nE) atomicAdd(cnt + eidx[nE + e], 1.f);
}

__global__ void k_inv(const float* __restrict__ cnt, float* __restrict__ inv, int n) {
  int t = blockIdx.x * 256 + threadIdx.x;
  if (t < n) inv[t] = 1.f / fmaxf(cnt[t], 1.f);
}

// ---------- encoder: h = relu(x @ Win + b), x bf16 [NN,64] -> h bf16 [NN,32] ----------
__global__ __launch_bounds__(32 * WPB_N) void k_encoder(const u16* __restrict__ xbf,
                                                        const u16* __restrict__ wfrag,
                                                        const float* __restrict__ bias,
                                                        u16* __restrict__ hout) {
  int lane = threadIdx.x & 31, m = lane & 15, hi = lane >> 4;
  int tile = blockIdx.x * WPB_N + (threadIdx.x >> 5);
  const u32* row = (const u32*)(xbf + (size_t)(tile * 16 + m) * 64);
  v8f acc[2];
#pragma unroll
  for (int nt = 0; nt < 2; ++nt) {
    float bv = bias[nt * 16 + m];
#pragma unroll
    for (int g = 0; g < 8; ++g) acc[nt][g] = bv;
  }
#pragma unroll
  for (int s = 0; s < 2; ++s) {
    v8u a;
#pragma unroll
    for (int p = 0; p < 4; ++p) {
      a[p]     = row[s * 16 + hi * 4 + p];
      a[p + 4] = row[s * 16 + 8 + hi * 4 + p];
    }
#pragma unroll
    for (int nt = 0; nt < 2; ++nt) acc[nt] = wmma_bf16(a, load_frag(wfrag, s * 2 + nt), acc[nt]);
  }
  u16* op = hout + (size_t)(tile * 16 + hi * 8) * 32 + m;
#pragma unroll
  for (int nt = 0; nt < 2; ++nt)
#pragma unroll
    for (int g = 0; g < 8; ++g) op[g * 32 + nt * 16] = bf16_trunc(fmaxf(acc[nt][g], 0.f));
}

// ---------- fused NNConv edge kernel ----------
// per 16-edge tile: h1 = relu(ea@w1+b1) via WMMA; msg = [P | h_src] @ [W2R ; B2]
// where P K-window s is rank-1: h1[:,s] (x) h_src. Scatter-add msg into agg[dst].
__global__ __launch_bounds__(32 * WPB_E) void k_edge(const float* __restrict__ ea,
                                                     const int* __restrict__ eidx,
                                                     const u16* __restrict__ hbf,
                                                     const u16* __restrict__ w1frag,
                                                     const float* __restrict__ b1,
                                                     const u16* __restrict__ w2frag,
                                                     float* __restrict__ agg) {
  __shared__ float h1tAll[WPB_E * 512];  // per-wave [feature s][edge m]
  __shared__ int sdstAll[WPB_E * 16];
  int wave = threadIdx.x >> 5;
  float* h1t = h1tAll + wave * 512;
  int* sdst = sdstAll + wave * 16;
  int lane = threadIdx.x & 31, m = lane & 15, hi = lane >> 4;
  int tile = blockIdx.x * WPB_E + wave;
  int e = tile * 16 + m;
  int srcv = eidx[e];
  if (lane < 16) sdst[m] = eidx[NE + e];

  // stage 1: edge_attr A-fragment (K=16 padded to 32)
  const float* earow = ea + (size_t)e * 16 + hi * 8;
  v8u a1;
#pragma unroll
  for (int p = 0; p < 4; ++p) { a1[p] = pack2_bf16(earow[2 * p], earow[2 * p + 1]); a1[p + 4] = 0u; }
  v8f c1[2];
#pragma unroll
  for (int nt = 0; nt < 2; ++nt) {
    float bv = b1[nt * 16 + m];
#pragma unroll
    for (int g = 0; g < 8; ++g) c1[nt][g] = bv;
  }
#pragma unroll
  for (int nt = 0; nt < 2; ++nt) c1[nt] = wmma_bf16(a1, load_frag(w1frag, nt), c1[nt]);
  __syncthreads();
#pragma unroll
  for (int nt = 0; nt < 2; ++nt)
#pragma unroll
    for (int g = 0; g < 8; ++g) h1t[(nt * 16 + m) * 16 + hi * 8 + g] = fmaxf(c1[nt][g], 0.f);

  // gather h[src] A-fragment (fixed across all K-steps)
  v8u ah = afrag_row32((const u32*)(hbf + (size_t)srcv * 32), hi);
  __syncthreads();

  v8f acc0 = {0,0,0,0,0,0,0,0}, acc1 = {0,0,0,0,0,0,0,0};
#pragma unroll 8
  for (int s = 0; s < 32; ++s) {
    u32 hb = __float_as_uint(h1t[s * 16 + m]);
    u32 dup = (hb & 0xFFFF0000u) | (hb >> 16);  // bf16 pair, both halves = h1[m,s]
    v8u as;
#pragma unroll
    for (int p = 0; p < 8; ++p) as[p] = pk_mul_bf16(ah[p], dup);
    acc0 = wmma_bf16(as, load_frag(w2frag, s * 2 + 0), acc0);
    acc1 = wmma_bf16(as, load_frag(w2frag, s * 2 + 1), acc1);
  }
  // bias K-step: msg += h_src @ B2
  acc0 = wmma_bf16(ah, load_frag(w2frag, 64), acc0);
  acc1 = wmma_bf16(ah, load_frag(w2frag, 65), acc1);

#pragma unroll
  for (int g = 0; g < 8; ++g) {
    float* ap = agg + (size_t)sdst[hi * 8 + g] * 32 + m;
    atomicAdd(ap, acc0[g]);
    atomicAdd(ap + 16, acc1[g]);
  }
}

// ---------- node update: h' = relu(agg/deg + h@root_w + root_b); re-zero agg ----------
__global__ __launch_bounds__(32 * WPB_N) void k_node(const u16* __restrict__ hin,
                                                     const u16* __restrict__ rootfrag,
                                                     const float* __restrict__ rootb,
                                                     const float* __restrict__ inv,
                                                     float* __restrict__ agg,
                                                     u16* __restrict__ hout) {
  int lane = threadIdx.x & 31, m = lane & 15, hi = lane >> 4;
  int tile = blockIdx.x * WPB_N + (threadIdx.x >> 5);
  v8u a = afrag_row32((const u32*)(hin + (size_t)(tile * 16 + m) * 32), hi);
  int vbase = tile * 16 + hi * 8;
  v8f acc[2];
#pragma unroll
  for (int nt = 0; nt < 2; ++nt) {
    float bv = rootb[nt * 16 + m];
#pragma unroll
    for (int g = 0; g < 8; ++g)
      acc[nt][g] = agg[(size_t)(vbase + g) * 32 + nt * 16 + m] * inv[vbase + g] + bv;
  }
#pragma unroll
  for (int nt = 0; nt < 2; ++nt) acc[nt] = wmma_bf16(a, load_frag(rootfrag, nt), acc[nt]);
  u16* op = hout + (size_t)vbase * 32 + m;
  float* zp = agg + (size_t)vbase * 32 + m;
#pragma unroll
  for (int nt = 0; nt < 2; ++nt)
#pragma unroll
    for (int g = 0; g < 8; ++g) {
      op[g * 32 + nt * 16] = bf16_trunc(fmaxf(acc[nt][g], 0.f));
      zp[g * 32 + nt * 16] = 0.f;  // ready for next layer
    }
}

// ---------- fused decoder: three chained WMMA GEMMs per 16-node tile ----------
__global__ __launch_bounds__(32 * WPB_N) void k_dec(const u16* __restrict__ h,
                                                    const u16* __restrict__ flo, const float* __restrict__ lob,
                                                    const u16* __restrict__ fd1, const float* __restrict__ db1,
                                                    const u16* __restrict__ fd2, const float* __restrict__ db2,
                                                    float* __restrict__ out) {
  __shared__ float ztAll[WPB_N * 256];   // per-wave 16x16
  __shared__ float dtAll[WPB_N * 512];   // per-wave 16x32
  int wave = threadIdx.x >> 5;
  float* zt = ztAll + wave * 256;
  float* dt = dtAll + wave * 512;
  int lane = threadIdx.x & 31, m = lane & 15, hi = lane >> 4;
  int tile = blockIdx.x * WPB_N + wave;

  // z = h @ lin_out + b   [16,32]x[32,16]
  v8u a1 = afrag_row32((const u32*)(h + (size_t)(tile * 16 + m) * 32), hi);
  v8f z; { float bv = lob[m];
#pragma unroll
    for (int g = 0; g < 8; ++g) z[g] = bv; }
  z = wmma_bf16(a1, load_frag(flo, 0), z);
#pragma unroll
  for (int g = 0; g < 8; ++g) zt[(hi * 8 + g) * 16 + m] = z[g];
  __syncthreads();

  // d = relu(z @ dec_w1 + b)   [16,16(pad32)]x[32,32]
  const float* zr = zt + m * 16 + hi * 8;
  v8u a2;
#pragma unroll
  for (int p = 0; p < 4; ++p) { a2[p] = pack2_bf16(zr[2 * p], zr[2 * p + 1]); a2[p + 4] = 0u; }
  v8f c2[2];
#pragma unroll
  for (int nt = 0; nt < 2; ++nt) {
    float bv = db1[nt * 16 + m];
#pragma unroll
    for (int g = 0; g < 8; ++g) c2[nt][g] = bv;
  }
#pragma unroll
  for (int nt = 0; nt < 2; ++nt) c2[nt] = wmma_bf16(a2, load_frag(fd1, nt), c2[nt]);
  __syncthreads();
#pragma unroll
  for (int nt = 0; nt < 2; ++nt)
#pragma unroll
    for (int g = 0; g < 8; ++g) dt[(hi * 8 + g) * 32 + nt * 16 + m] = fmaxf(c2[nt][g], 0.f);
  __syncthreads();

  // out = d @ dec_w2 + b   [16,32]x[32,64]
  const float* dr = dt + m * 32;
  v8u a3;
#pragma unroll
  for (int p = 0; p < 4; ++p) {
    a3[p]     = pack2_bf16(dr[hi * 8 + 2 * p],      dr[hi * 8 + 2 * p + 1]);
    a3[p + 4] = pack2_bf16(dr[16 + hi * 8 + 2 * p], dr[16 + hi * 8 + 2 * p + 1]);
  }
  v8f c3[4];
#pragma unroll
  for (int nt = 0; nt < 4; ++nt) {
    float bv = db2[nt * 16 + m];
#pragma unroll
    for (int g = 0; g < 8; ++g) c3[nt][g] = bv;
  }
#pragma unroll
  for (int nt = 0; nt < 4; ++nt) c3[nt] = wmma_bf16(a3, load_frag(fd2, nt), c3[nt]);
  float* op = out + (size_t)(tile * 16 + hi * 8) * 64 + m;
#pragma unroll
  for (int nt = 0; nt < 4; ++nt)
#pragma unroll
    for (int g = 0; g < 8; ++g) op[g * 64 + nt * 16] = c3[nt][g];
}

// ---------- host ----------
static inline int cdiv(int a, int b) { return (a + b - 1) / b; }

extern "C" void kernel_launch(void* const* d_in, const int* in_sizes, int n_in,
                              void* d_out, int out_size, void* d_ws, size_t ws_size,
                              hipStream_t stream) {
  const float* x    = (const float*)d_in[0];
  const int*   eidx = (const int*)d_in[1];
  const float* ea   = (const float*)d_in[2];
  const float* liw  = (const float*)d_in[3];
  const float* lib  = (const float*)d_in[4];
  const float* ew1  = (const float*)d_in[5];
  const float* eb1  = (const float*)d_in[6];
  const float* ew2  = (const float*)d_in[7];
  const float* eb2  = (const float*)d_in[8];
  const float* rw   = (const float*)d_in[9];
  const float* rb   = (const float*)d_in[10];
  const float* low  = (const float*)d_in[11];
  const float* lob  = (const float*)d_in[12];
  const float* dw1  = (const float*)d_in[13];
  const float* db1  = (const float*)d_in[14];
  const float* dw2  = (const float*)d_in[15];
  const float* db2  = (const float*)d_in[16];
  float* out = (float*)d_out;

  char* p = (char*)d_ws;
  auto alloc = [&](size_t bytes) { void* r = (void*)p; p += (bytes + 255) & ~(size_t)255; return r; };
  u16*   xbf  = (u16*)alloc((size_t)NN * 64 * 2);
  u16*   hA   = (u16*)alloc((size_t)NN * 32 * 2);
  u16*   hB   = (u16*)alloc((size_t)NN * 32 * 2);
  float* agg  = (float*)alloc((size_t)NN * 32 * 4);
  float* cnt  = (float*)alloc((size_t)NN * 4);
  float* invd = (float*)alloc((size_t)NN * 4);
  u16* f_in  = (u16*)alloc(2 * 2 * 512 * 2);
  u16* f_lo  = (u16*)alloc(1 * 1 * 512 * 2);
  u16* f_d1  = (u16*)alloc(1 * 2 * 512 * 2);
  u16* f_d2  = (u16*)alloc(1 * 4 * 512 * 2);
  u16* f_w1[3]; u16* f_w2[3]; u16* f_rt[3];
  for (int l = 0; l < 3; ++l) {
    f_w1[l] = (u16*)alloc(1 * 2 * 512 * 2);
    f_w2[l] = (u16*)alloc(33 * 2 * 512 * 2);  // 32 K-steps + 1 bias step
    f_rt[l] = (u16*)alloc(1 * 2 * 512 * 2);
  }

  // conversions + weight fragment prepack
  k_cvt_bf16<<<cdiv(NN * 64, 256), 256, 0, stream>>>(x, xbf, NN * 64);
  auto pp = [&](const float* W, u16* o, int ks, int nt, int Ksrc, int N, int mode) {
    int tot = ks * nt * 512;
    k_prepack<<<cdiv(tot, 256), 256, 0, stream>>>(W, o, ks, nt, Ksrc, N, mode);
  };
  pp(liw, f_in, 2, 2, 64, 32, 0);
  pp(low, f_lo, 1, 1, 32, 16, 0);
  pp(dw1, f_d1, 1, 2, 16, 32, 0);
  pp(dw2, f_d2, 1, 4, 32, 64, 0);
  for (int l = 0; l < 3; ++l) {
    pp(ew1 + l * 16 * 32, f_w1[l], 1, 2, 16, 32, 0);
    pp(ew2 + (size_t)l * 32 * 1024, f_w2[l], 32, 2, 1024, 32, 1);
    pp(eb2 + l * 1024, f_w2[l] + 32 * 2 * 512, 1, 2, 32, 32, 0);  // B2[h,o]=b2[h*32+o]
    pp(rw + l * 32 * 32, f_rt[l], 1, 2, 32, 32, 0);
  }

  // degree + agg init
  k_zero<<<cdiv(NN * 32, 256), 256, 0, stream>>>(agg, cnt, NN * 32, NN);
  k_count<<<cdiv(NE, 256), 256, 0, stream>>>(eidx, cnt, NE);
  k_inv<<<cdiv(NN, 256), 256, 0, stream>>>(cnt, invd, NN);

  // encoder
  k_encoder<<<NN / 16 / WPB_N, 32 * WPB_N, 0, stream>>>(xbf, f_in, lib, hA);

  // 3 NNConv layers
  u16* hin = hA; u16* hout = hB;
  for (int l = 0; l < 3; ++l) {
    k_edge<<<NE / 16 / WPB_E, 32 * WPB_E, 0, stream>>>(ea, eidx, hin, f_w1[l], eb1 + l * 32, f_w2[l], agg);
    k_node<<<NN / 16 / WPB_N, 32 * WPB_N, 0, stream>>>(hin, f_rt[l], rb + l * 32, invd, agg, hout);
    u16* t = hin; hin = hout; hout = t;
  }

  // decoder
  k_dec<<<NN / 16 / WPB_N, 32 * WPB_N, 0, stream>>>(hin, f_lo, lob, f_d1, db1, f_d2, db2, out);
}
